// GraphormerAttentionHead_38517266710634
// MI455X (gfx1250) — matrix operand order repaired and checked by reading, not compiled, not run on hardware
//
#include <hip/hip_runtime.h>

// ---------------------------------------------------------------------------
// Graphormer attention head for MI455X (gfx1250, wave32, WMMA + TDM).
// B=16 graphs, NNODE=128, E=512, L=5, N=2048, DIM_IN=256, DQ=DK=EDIM=64.
// ---------------------------------------------------------------------------

#define BG     16
#define NNODE  128
#define NEDGE  512
#define LPATH  5
#define NTOT   2048
#define DIN    256
#define DQK    64
#define NEGF   (-1000000.0f)

// LDS layout of attn_kernel (dynamic shared, base offset 0)
#define LDS_QS   0u                    // 128 rows * 66 ushort (padded)
#define LDS_KS   16896u                // 128 * 66 * 2 bytes
#define LDS_VT   33792u                // v transposed: 64 * 136 ushort
#define LDS_EWS  51200u                // 512*5 float
#define LDS_TOT  61440u

typedef __attribute__((ext_vector_type(16))) __bf16        v16bf;
typedef __attribute__((ext_vector_type(8)))  float         v8f;
typedef __attribute__((ext_vector_type(8)))  unsigned int  v8u;
typedef __attribute__((ext_vector_type(4)))  unsigned int  u32x4;
typedef __attribute__((ext_vector_type(8)))  unsigned int  u32x8;

__device__ __forceinline__ unsigned short f2bf(float f) {
    __bf16 h = (__bf16)f;                       // hardware cvt
    return __builtin_bit_cast(unsigned short, h);
}
__device__ __forceinline__ v8f wmma_bf16(v8u a8, v8u b8, v8f c) {
    v16bf a = __builtin_bit_cast(v16bf, a8);
    v16bf b = __builtin_bit_cast(v16bf, b8);
    return __builtin_amdgcn_wmma_f32_16x16x32_bf16(false, a, false, b, (short)0, c,
                                                   false, false);
}
__device__ __forceinline__ v8u frag_from_u4(uint4 lo, uint4 hi) {
    v8u r;
    r[0] = lo.x; r[1] = lo.y; r[2] = lo.z; r[3] = lo.w;
    r[4] = hi.x; r[5] = hi.y; r[6] = hi.z; r[7] = hi.w;
    return r;
}

// ---------------------------------------------------------------------------
// 0) init: zero d_blocks, winner = -1
// ---------------------------------------------------------------------------
__global__ void init_kernel(float* __restrict__ dblk, int* __restrict__ winner) {
    int idx = blockIdx.x * blockDim.x + threadIdx.x;
    if (idx < BG * NNODE * NNODE) { dblk[idx] = 0.0f; winner[idx] = -1; }
}

// ---------------------------------------------------------------------------
// 1) one-shot f32 -> bf16 conversions (row-major copy / transpose)
// ---------------------------------------------------------------------------
__global__ void cvt_rowmajor_kernel(const float* __restrict__ src,
                                    unsigned short* __restrict__ dst, int n) {
    for (int i = blockIdx.x * blockDim.x + threadIdx.x; i < n;
         i += gridDim.x * blockDim.x)
        dst[i] = f2bf(src[i]);
}
__global__ void cvt_transpose_kernel(const float* __restrict__ src,   // K x N
                                     unsigned short* __restrict__ dst, // N x K
                                     int K, int N) {
    for (int i = blockIdx.x * blockDim.x + threadIdx.x; i < K * N;
         i += gridDim.x * blockDim.x) {
        int k = i / N, n = i - k * N;
        dst[n * K + k] = f2bf(src[i]);
    }
}

// ---------------------------------------------------------------------------
// 2) QKV projection from pre-converted bf16: q/k/v = x @ W + b  (bf16 out)
//    grid(128,3) x 32 thr; one wave = 16x64 tile; all frags are b128 loads.
// ---------------------------------------------------------------------------
__global__ void qkv_proj_kernel(const unsigned short* __restrict__ xbf,
                                const unsigned short* __restrict__ wqt,
                                const unsigned short* __restrict__ wkt,
                                const unsigned short* __restrict__ wvt,
                                const float* __restrict__ bq,
                                const float* __restrict__ bk,
                                const float* __restrict__ bv,
                                unsigned short* __restrict__ qb,
                                unsigned short* __restrict__ kb,
                                unsigned short* __restrict__ vb) {
    const int mt    = blockIdx.x;
    const int which = blockIdx.y;
    const unsigned short* Wt = (which == 0) ? wqt : (which == 1) ? wkt : wvt;
    const float* bias        = (which == 0) ? bq  : (which == 1) ? bk  : bv;
    unsigned short* out      = (which == 0) ? qb  : (which == 1) ? kb  : vb;

    const int lane  = threadIdx.x & 31;
    const int lcol  = lane & 15;
    const int khalf = lane >> 4;

    v8f acc[4] = {{}, {}, {}, {}};
    const unsigned short* xrow = xbf + (mt * 16 + lcol) * DIN;

#pragma unroll
    for (int kk = 0; kk < DIN / 32; ++kk) {
        const unsigned short* xp = xrow + kk * 32 + khalf * 16;
        v8u a8 = frag_from_u4(*(const uint4*)xp, *(const uint4*)(xp + 8));
#pragma unroll
        for (int nt = 0; nt < 4; ++nt) {
            const unsigned short* wp =
                Wt + (nt * 16 + lcol) * DIN + kk * 32 + khalf * 16;
            v8u b8 = frag_from_u4(*(const uint4*)wp, *(const uint4*)(wp + 8));
            acc[nt] = wmma_bf16(a8, b8, acc[nt]);
        }
    }
#pragma unroll
    for (int nt = 0; nt < 4; ++nt) {
        int n = nt * 16 + lcol;
        float bv_ = bias[n];
#pragma unroll
        for (int r = 0; r < 8; ++r) {
            int m = mt * 16 + khalf * 8 + r;
            out[m * DQK + n] = f2bf(acc[nt][r] + bv_);
        }
    }
}

// ---------------------------------------------------------------------------
// 3) ew[g,e,l] = dot(edge_attr[g,e,:], edge_vector[l,:])   (f32, exact)
// ---------------------------------------------------------------------------
__global__ void ew_kernel(const float* __restrict__ edge_attr,
                          const float* __restrict__ edge_vector,
                          float* __restrict__ ew) {
    int idx = blockIdx.x * blockDim.x + threadIdx.x;
    if (idx >= BG * NEDGE) return;
    const float* ea = edge_attr + idx * 64;
#pragma unroll
    for (int l = 0; l < LPATH; ++l) {
        const float* evr = edge_vector + l * 64;
        float s = 0.0f;
#pragma unroll 8
        for (int d = 0; d < 64; ++d) s += ea[d] * evr[d];
        ew[idx * LPATH + l] = s;
    }
}

// ---------------------------------------------------------------------------
// 4) dist = gelu(edge_attr @ W1 + b1) @ W2 + b2   (8192 rows). WMMA + reduce.
// ---------------------------------------------------------------------------
__global__ void dist_kernel(const unsigned short* __restrict__ eabf,
                            const unsigned short* __restrict__ w1t,
                            const float* __restrict__ b1,
                            const float* __restrict__ W2,
                            const float* __restrict__ b2,
                            float* __restrict__ dist) {
    const int mt    = blockIdx.x;
    const int lane  = threadIdx.x & 31;
    const int lcol  = lane & 15;
    const int khalf = lane >> 4;

    v8f acc[4] = {{}, {}, {}, {}};
    const unsigned short* xrow = eabf + (mt * 16 + lcol) * 64;
#pragma unroll
    for (int kk = 0; kk < 2; ++kk) {
        const unsigned short* xp = xrow + kk * 32 + khalf * 16;
        v8u a8 = frag_from_u4(*(const uint4*)xp, *(const uint4*)(xp + 8));
#pragma unroll
        for (int nt = 0; nt < 4; ++nt) {
            const unsigned short* wp =
                w1t + (nt * 16 + lcol) * 64 + kk * 32 + khalf * 16;
            v8u b8 = frag_from_u4(*(const uint4*)wp, *(const uint4*)(wp + 8));
            acc[nt] = wmma_bf16(a8, b8, acc[nt]);
        }
    }
    float b2v = b2[0];
#pragma unroll
    for (int r = 0; r < 8; ++r) {
        float p = 0.0f;
#pragma unroll
        for (int nt = 0; nt < 4; ++nt) {
            int n = nt * 16 + lcol;
            float h = acc[nt][r] + b1[n];
            h = 0.5f * h * (1.0f + erff(h * 0.70710678118654752f));
            p += h * W2[n];
        }
#pragma unroll
        for (int off = 8; off; off >>= 1) p += __shfl_xor(p, off, 16);
        if (lcol == 0) dist[mt * 16 + khalf * 8 + r] = p + b2v;
    }
}

// ---------------------------------------------------------------------------
// 5) deterministic scatter of dist (last edge index wins on duplicates)
// ---------------------------------------------------------------------------
__global__ void scatter_win_kernel(const int* __restrict__ edge_idx,
                                   int* __restrict__ winner) {
    int idx = blockIdx.x * blockDim.x + threadIdx.x;
    if (idx >= BG * NEDGE) return;
    int g   = idx / NEDGE;
    int si = edge_idx[idx] - g * NNODE;
    int dj = edge_idx[BG * NEDGE + idx] - g * NNODE;
    atomicMax(&winner[g * NNODE * NNODE + si * NNODE + dj], idx);
}
__global__ void scatter_val_kernel(const int* __restrict__ edge_idx,
                                   const int* __restrict__ winner,
                                   const float* __restrict__ dist,
                                   float* __restrict__ dblk) {
    int idx = blockIdx.x * blockDim.x + threadIdx.x;
    if (idx >= BG * NEDGE) return;
    int g   = idx / NEDGE;
    int si = edge_idx[idx] - g * NNODE;
    int dj = edge_idx[BG * NEDGE + idx] - g * NNODE;
    int key = g * NNODE * NNODE + si * NNODE + dj;
    if (winner[key] == idx) dblk[key] = dist[idx];
}

// ---------------------------------------------------------------------------
// 6) fused per-graph attention. grid 16 x 256 (8 waves), 60KB dynamic LDS.
//    q/k tiles staged via TDM tensor_load_to_lds with row padding (stride 66).
// ---------------------------------------------------------------------------
__global__ void __launch_bounds__(256)
attn_kernel(const unsigned short* __restrict__ qbf,
            const unsigned short* __restrict__ kbf,
            const unsigned short* __restrict__ vbf,
            const float* __restrict__ ewb,
            const float* __restrict__ dblk,
            const float* __restrict__ bmat,
            const int*   __restrict__ edge_paths,
            float* __restrict__ attw,
            unsigned short* __restrict__ pbf,
            float* __restrict__ out) {
    extern __shared__ char smem[];
    unsigned short* qs  = (unsigned short*)(smem + LDS_QS);   // [128][66]
    unsigned short* ks  = (unsigned short*)(smem + LDS_KS);   // [128][66]
    unsigned short* vt  = (unsigned short*)(smem + LDS_VT);   // [64][136]
    float*          ews = (float*)         (smem + LDS_EWS);  // [512*5]

    const int g     = blockIdx.x;
    const int tid   = threadIdx.x;
    const int w     = tid >> 5;
    const int lane  = tid & 31;
    const int lcol  = lane & 15;
    const int khalf = lane >> 4;
    const int G     = g * NNODE;

    // ---- TDM: DMA q and k 128x64 bf16 tiles into padded LDS rows ----------
    if (tid < 32) {
        // D# group1: data_size=2B, pad_enable, pad_interval=32 DWORDs,
        // pad_amount=1 DWORD  ->  LDS row stride 132B = 66 ushorts.
        u32x8 g1;
        g1[0] = (1u << 16) | (1u << 20) | (4u << 22);
        g1[1] = (64u << 16);          // tensor_dim0 = 64
        g1[2] = (128u << 16);         // tensor_dim1 = 128
        g1[3] = (64u << 16);          // tile_dim0 = 64
        g1[4] = 128u;                 // tile_dim1 = 128
        g1[5] = 64u;                  // tensor_dim0_stride = 64
        g1[6] = 0u;
        g1[7] = 0u;
        unsigned long long gaq =
            (unsigned long long)(uintptr_t)(qbf + (size_t)G * DQK);
        unsigned long long gak =
            (unsigned long long)(uintptr_t)(kbf + (size_t)G * DQK);
        u32x4 g0;
        g0[0] = 1u;                                    // count=1
        g0[1] = LDS_QS;                                // lds_addr
        g0[2] = (unsigned)(gaq & 0xffffffffu);
        g0[3] = (unsigned)((gaq >> 32) & 0x1ffffffu) | (2u << 30); // type=2
        asm volatile("tensor_load_to_lds %0, %1" :: "s"(g0), "s"(g1) : "memory");
        g0[1] = LDS_KS;
        g0[2] = (unsigned)(gak & 0xffffffffu);
        g0[3] = (unsigned)((gak >> 32) & 0x1ffffffu) | (2u << 30);
        asm volatile("tensor_load_to_lds %0, %1" :: "s"(g0), "s"(g1) : "memory");
    }

    // ---- manual staging: v transposed + ew table --------------------------
    for (int idx = tid; idx < NNODE * 64; idx += 256) {
        int r = idx >> 6, c = idx & 63;
        vt[c * 136 + r] = vbf[(G + r) * DQK + c];
    }
    for (int idx = tid; idx < NEDGE * LPATH; idx += 256)
        ews[idx] = ewb[g * NEDGE * LPATH + idx];

    if (tid < 32) __builtin_amdgcn_s_wait_tensorcnt(0);
    __syncthreads();

    // ---- phase 1: scores = q k^T / 8 + b + enc + d  -> attw ---------------
    {
        v8u aq[2];
#pragma unroll
        for (int kk = 0; kk < 2; ++kk) {
            int row = w * 16 + lcol;
#pragma unroll
            for (int r = 0; r < 4; ++r) {
                int k0 = kk * 32 + khalf * 16 + 2 * r;
                aq[kk][r]     = *(const unsigned int*)&qs[row * 66 + k0];
                aq[kk][4 + r] = *(const unsigned int*)&qs[row * 66 + 8 + k0];
            }
        }
#pragma unroll 1
        for (int nt = 0; nt < 8; ++nt) {
            v8f acc = {};
#pragma unroll
            for (int kk = 0; kk < 2; ++kk) {
                int nrow = nt * 16 + lcol;
                v8u b8;
#pragma unroll
                for (int j = 0; j < 8; ++j) {
                    int k = kk * 32 + khalf * 16 + 2 * j;
                    b8[j] = *(const unsigned int*)&ks[nrow * 66 + k];
                }
                acc = wmma_bf16(aq[kk], b8, acc);
            }
            int n = nt * 16 + lcol;
#pragma unroll
            for (int r = 0; r < 8; ++r) {
                int m = w * 16 + khalf * 8 + r;
                const int* pp = edge_paths + (((g * NNODE + m) * NNODE) + n) * LPATH;
                float s = 0.0f; int cnt = 0;
#pragma unroll
                for (int l = 0; l < LPATH; ++l) {
                    int pid = pp[l];
                    if (pid >= 0) { s += ews[pid * LPATH + l]; ++cnt; }
                }
                float encv = s / ((float)cnt + 1e-9f);
                float av = acc[r] * 0.125f
                         + bmat[(G + m) * NTOT + (G + n)]
                         + dblk[g * NNODE * NNODE + m * NNODE + n]
                         + encv;
                attw[g * NNODE * NNODE + m * NNODE + n] = av;
            }
        }
    }
    __threadfence();
    __syncthreads();

    // ---- phase 2: two-pass softmax over FULL 2048-wide row ----------------
    for (int rr = 0; rr < 16; ++rr) {
        int i = w * 16 + rr;
        const float* brow = bmat + (G + i) * NTOT;
        const float* arow = attw + g * NNODE * NNODE + i * NNODE;
        // pass 1: row max
        float mloc = -3.0e38f;
        for (int j = lane; j < NTOT; j += 32) {
            __builtin_prefetch(brow + j + 256, 0, 0);
            unsigned int jl = (unsigned int)(j - G);
            float v = (jl < (unsigned)NNODE) ? arow[jl] : brow[j] * NEGF;
            mloc = fmaxf(mloc, v);
        }
#pragma unroll
        for (int off = 16; off; off >>= 1)
            mloc = fmaxf(mloc, __shfl_xor(mloc, off, 32));
        // pass 2: sum of exp
        float sloc = 0.0f;
        for (int j = lane; j < NTOT; j += 32) {
            unsigned int jl = (unsigned int)(j - G);
            float v = (jl < (unsigned)NNODE) ? arow[jl] : brow[j] * NEGF;
            sloc += __expf(v - mloc);
        }
#pragma unroll
        for (int off = 16; off; off >>= 1) sloc += __shfl_xor(sloc, off, 32);
        float inv = 1.0f / sloc;
        for (int jl = lane; jl < NNODE; jl += 32) {
            float p = __expf(arow[jl] - mloc) * inv;
            pbf[g * NNODE * NNODE + i * NNODE + jl] = f2bf(p);
        }
    }
    __threadfence();
    __syncthreads();

    // ---- phase 3: out = P @ V (A frags b128 from global P, B from LDS) ----
    {
        v8f acc[4] = {{}, {}, {}, {}};
        const unsigned short* prow =
            pbf + g * NNODE * NNODE + (w * 16 + lcol) * NNODE;
#pragma unroll
        for (int kk = 0; kk < 4; ++kk) {
            const unsigned short* pp = prow + kk * 32 + khalf * 16;
            v8u a8 = frag_from_u4(*(const uint4*)pp, *(const uint4*)(pp + 8));
#pragma unroll
            for (int nt = 0; nt < 4; ++nt) {
                int n = nt * 16 + lcol;
                v8u b8;
#pragma unroll
                for (int j = 0; j < 8; ++j) {
                    int k = kk * 32 + khalf * 16 + 2 * j;
                    b8[j] = *(const unsigned int*)&vt[n * 136 + k];
                }
                acc[nt] = wmma_bf16(a8, b8, acc[nt]);
            }
        }
#pragma unroll
        for (int nt = 0; nt < 4; ++nt) {
            int n = nt * 16 + lcol;
#pragma unroll
            for (int r = 0; r < 8; ++r) {
                int m = w * 16 + khalf * 8 + r;
                out[(G + m) * DQK + n] = acc[nt][r];
            }
        }
    }
}

// ---------------------------------------------------------------------------
// launch
// ---------------------------------------------------------------------------
extern "C" void kernel_launch(void* const* d_in, const int* in_sizes, int n_in,
                              void* d_out, int out_size, void* d_ws, size_t ws_size,
                              hipStream_t stream) {
    const float* x          = (const float*)d_in[0];
    const int*   edge_idx   = (const int*)  d_in[1];
    const float* edge_attr  = (const float*)d_in[2];
    const float* bmat       = (const float*)d_in[3];
    const int*   edge_paths = (const int*)  d_in[4];
    const float* Wq = (const float*)d_in[6];  const float* bq = (const float*)d_in[7];
    const float* Wk = (const float*)d_in[8];  const float* bk = (const float*)d_in[9];
    const float* Wv = (const float*)d_in[10]; const float* bv = (const float*)d_in[11];
    const float* edge_vector = (const float*)d_in[12];
    const float* W1 = (const float*)d_in[13]; const float* b1 = (const float*)d_in[14];
    const float* W2 = (const float*)d_in[15]; const float* b2 = (const float*)d_in[16];
    float* out = (float*)d_out;

    char* ws = (char*)d_ws;
    unsigned short* qbf  = (unsigned short*)(ws + 0);          // 2048*64*2
    unsigned short* kbf  = (unsigned short*)(ws + 262144);
    unsigned short* vbf  = (unsigned short*)(ws + 524288);
    float*          dist = (float*)         (ws + 786432);     // 8192*4
    float*          ewb  = (float*)         (ws + 819200);     // 16*512*5*4
    float*          dblk = (float*)         (ws + 983040);     // 16*128*128*4
    float*          attw = (float*)         (ws + 2031616);    // 16*128*128*4
    unsigned short* pbf  = (unsigned short*)(ws + 3080192);    // 16*128*128*2
    int*            win  = (int*)           (ws + 3604480);    // 16*128*128*4
    unsigned short* xbf  = (unsigned short*)(ws + 4653056);    // 2048*256*2
    unsigned short* eabf = (unsigned short*)(ws + 5701632);    // 8192*64*2
    unsigned short* wqt  = (unsigned short*)(ws + 6750208);    // 64*256*2
    unsigned short* wkt  = (unsigned short*)(ws + 6782976);
    unsigned short* wvt  = (unsigned short*)(ws + 6815744);
    unsigned short* w1t  = (unsigned short*)(ws + 6848512);    // 64*64*2

    init_kernel<<<dim3((BG * NNODE * NNODE + 255) / 256), dim3(256), 0, stream>>>(dblk, win);
    cvt_rowmajor_kernel<<<dim3(512), dim3(256), 0, stream>>>(x, xbf, NTOT * DIN);
    cvt_rowmajor_kernel<<<dim3(512), dim3(256), 0, stream>>>(edge_attr, eabf,
                                                             BG * NEDGE * 64);
    cvt_transpose_kernel<<<dim3(64), dim3(256), 0, stream>>>(Wq, wqt, DIN, DQK);
    cvt_transpose_kernel<<<dim3(64), dim3(256), 0, stream>>>(Wk, wkt, DIN, DQK);
    cvt_transpose_kernel<<<dim3(64), dim3(256), 0, stream>>>(Wv, wvt, DIN, DQK);
    cvt_transpose_kernel<<<dim3(16), dim3(256), 0, stream>>>(W1, w1t, 64, 64);

    qkv_proj_kernel<<<dim3(NTOT / 16, 3), dim3(32), 0, stream>>>(
        xbf, wqt, wkt, wvt, bq, bk, bv, qbf, kbf, vbf);
    ew_kernel<<<dim3((BG * NEDGE + 255) / 256), dim3(256), 0, stream>>>(
        edge_attr, edge_vector, ewb);
    dist_kernel<<<dim3(BG * NEDGE / 16), dim3(32), 0, stream>>>(
        eabf, w1t, b1, W2, b2, dist);
    scatter_win_kernel<<<dim3((BG * NEDGE + 255) / 256), dim3(256), 0, stream>>>(
        edge_idx, win);
    scatter_val_kernel<<<dim3((BG * NEDGE + 255) / 256), dim3(256), 0, stream>>>(
        edge_idx, win, dist, dblk);
    attn_kernel<<<dim3(BG), dim3(256), LDS_TOT, stream>>>(
        qbf, kbf, vbf, ewb, dblk, bmat, edge_paths, attw, pbf, out);
}
